// TopologicalEncoder_60060822667826
// MI455X (gfx1250) — matrix operand order repaired and checked by reading, not compiled, not run on hardware
//
#include <hip/hip_runtime.h>
#include <math.h>

// ---------------------------------------------------------------------------
// Problem constants (fixed by the reference)
// ---------------------------------------------------------------------------
#define B_   32
#define T_   4096
#define D_   128
#define H_   256
#define KL_  64      // K_LIFT
#define DM_  1024    // D_MODEL
#define KEFF 64      // min(T, MAX_PROXY)
#define KBUD 16.0f   // selector budget

typedef __attribute__((ext_vector_type(16))) __bf16 v16bf;
typedef __attribute__((ext_vector_type(16))) short  v16s;
typedef __attribute__((ext_vector_type(8)))  float  v8f;

// fp32 -> bf16 (round-to-nearest-even), bit-level (known-good codegen)
__device__ __forceinline__ short f2bf_s(float f) {
  unsigned u = __builtin_bit_cast(unsigned, f);
  u = u + 0x7FFFu + ((u >> 16) & 1u);
  return (short)(u >> 16);
}

// Branchless fast tanh: (e^{2x}-1) * rcp(e^{2x}+1); v_exp_f32 + v_rcp_f32.
__device__ __forceinline__ float fast_tanh(float x) {
  const float xc = fminf(fmaxf(x, -15.f), 15.f);
  const float e  = __expf(2.f * xc);
  return (e - 1.f) * __builtin_amdgcn_rcpf(e + 1.f);
}

// ---------------------------------------------------------------------------
// Weight pre-pack: W[K,N] f32 row-major -> bf16 fragments in the exact
// per-lane WMMA B-operand layout (ISA 7.12.2, 16-bit B, K=32 step):
//   fragment f = nt*(K/32) + ks ; lane L holds col N = nt*16 + (L&15),
//   element e (0..15) holds K = ks*32 + (L>>4)*8 + (e&7) + (e>>3)*16.
// Main kernels then fetch a whole B fragment with one 32-byte vector load.
// ---------------------------------------------------------------------------
__global__ __launch_bounds__(256) void k_pack_b(
    const float* __restrict__ W, short* __restrict__ out, int K, int N)
{
  const int idx = blockIdx.x * 256 + threadIdx.x;     // one bf16 element
  if (idx >= K * N) return;
  const int e    = idx & 15;
  const int lane = (idx >> 4) & 31;
  const int frag = idx >> 9;
  const int ksn  = K >> 5;
  const int nt   = frag / ksn;
  const int ks   = frag - nt * ksn;
  const int k    = ks * 32 + (lane >> 4) * 8 + (e & 7) + (e >> 3) * 16;
  const int col  = nt * 16 + (lane & 15);
  out[idx] = f2bf_s(W[(size_t)k * N + col]);
}

__device__ __forceinline__ v16bf load_b_frag(const short* __restrict__ packed,
                                             int frag, int lane) {
  return ((const v16bf*)packed)[frag * 32 + lane];    // 32B aligned vector load
}

// ---------------------------------------------------------------------------
// Kernel 1: saliency[b,t] = softplus( sum_h tanh(x@W1 + b1)[h] * w_s[h] )
// One wave per 16-row tile; h never touches memory.
// ---------------------------------------------------------------------------
__global__ __launch_bounds__(256) void k_saliency(
    const float* __restrict__ x, const short* __restrict__ W1p,
    const float* __restrict__ b1, const float* __restrict__ w_s,
    float* __restrict__ sal)
{
  const int wave = threadIdx.x >> 5;
  const int lane = threadIdx.x & 31;
  const int tile = blockIdx.x * 8 + wave;      // 8192 tiles of 16 rows
  const int row0 = tile * 16;
  const int m    = lane & 15;
  const int hi   = lane >> 4;

  // A-matrix: 16x128 slab of x as 4 k-steps of 16x32 bf16 (ISA 7.12.2)
  v16bf a[4];
  const float* xr = x + (size_t)(row0 + m) * D_;
  __builtin_prefetch(xr + 16 * D_, 0, 0);      // global_prefetch_b8: next tile
#pragma unroll
  for (int ks = 0; ks < 4; ++ks) {
    v16s s;
#pragma unroll
    for (int v = 0; v < 8; ++v) {
      int kb = ks * 32 + ((v & 4) ? 16 : 0) + hi * 8 + (v & 3) * 2;
      s[2 * v]     = f2bf_s(xr[kb]);
      s[2 * v + 1] = f2bf_s(xr[kb + 1]);
    }
    a[ks] = __builtin_bit_cast(v16bf, s);
  }

  float rp[8] = {0.f, 0.f, 0.f, 0.f, 0.f, 0.f, 0.f, 0.f};

#pragma unroll 1
  for (int nt = 0; nt < H_ / 16; ++nt) {       // 16 column tiles of H=256
    const int col = nt * 16 + m;
    v8f acc = {};
#pragma unroll
    for (int ks = 0; ks < 4; ++ks) {
      const v16bf bm = load_b_frag(W1p, nt * 4 + ks, lane);
      acc = __builtin_amdgcn_wmma_f32_16x16x32_bf16(
          false, a[ks], false, bm, (short)0, acc, false, false);
    }
    const float bb = b1[col], ws = w_s[col];
#pragma unroll
    for (int r = 0; r < 8; ++r) rp[r] += fast_tanh(acc[r] + bb) * ws;
  }

  // C layout: VGPR r = row (r + 8*hi), N = lane&15 -> reduce across 16 lanes
#pragma unroll
  for (int r = 0; r < 8; ++r)
    for (int msk = 8; msk >= 1; msk >>= 1) rp[r] += __shfl_xor(rp[r], msk, 32);

  if (m == 0) {
#pragma unroll
    for (int r = 0; r < 8; ++r) {
      const float s = rp[r];
      sal[row0 + r + 8 * hi] = (s > 20.f) ? s : log1pf(expf(s));
    }
  }
}

// ---------------------------------------------------------------------------
// Kernel 2: per batch row — cumsum(saliency), sigmoid selector, budget scale,
// refractory (roll -1), writes y_star directly into the output buffer.
// ---------------------------------------------------------------------------
__global__ __launch_bounds__(256) void k_selector(
    const float* __restrict__ sal, const float* __restrict__ log_temp,
    float* __restrict__ cums, float* __restrict__ ystar)
{
  __shared__ float s_sal[T_];
  __shared__ float s_y[T_];
  __shared__ float s_red[256];

  const int b = blockIdx.x, tid = threadIdx.x;
  const float* sr = sal + (size_t)b * T_;
  for (int i = tid; i < T_; i += 256) s_sal[i] = sr[i];
  __syncthreads();

  // ---- inclusive cumsum of saliency (chunk-of-16 + block scan) ----
  const int base = tid * 16;
  float loc[16], run = 0.f;
#pragma unroll
  for (int j = 0; j < 16; ++j) { run += s_sal[base + j]; loc[j] = run; }
  s_red[tid] = run;
  __syncthreads();
  for (int off = 1; off < 256; off <<= 1) {
    float t = (tid >= off) ? s_red[tid - off] : 0.f;
    __syncthreads();
    s_red[tid] += t;
    __syncthreads();
  }
  const float excl = s_red[tid] - run;
  float* cr = cums + (size_t)b * T_;
#pragma unroll
  for (int j = 0; j < 16; ++j) cr[base + j] = loc[j] + excl;

  // ---- sigmoid relaxation: y = sigmoid((sal - 0.5)/temp), y[0]=0 ----
  const float temp = fminf(fmaxf(expf(log_temp[0]), 0.1f), 10.0f);
  float ysum = 0.f;
#pragma unroll
  for (int j = 0; j < 16; ++j) {
    const int i = base + j;
    float y = 1.0f / (1.0f + expf(-((s_sal[i] - 0.5f) / temp)));
    if (i == 0) y = 0.f;
    s_y[i] = y;
    ysum += y;
  }
  __syncthreads();
  s_red[tid] = ysum;
  __syncthreads();
  for (int off = 128; off >= 1; off >>= 1) {
    if (tid < off) s_red[tid] += s_red[tid + off];
    __syncthreads();
  }
  const float budget = fmaxf(s_red[0], 1e-6f);
  const float scale  = fminf(KBUD / budget, 1.0f);

  // ---- refractory r=1: y *= min(2/(1+y[i]+y[i+1 mod T]), 1); y[0]=0 ----
  float* yo = ystar + (size_t)b * T_;
#pragma unroll
  for (int j = 0; j < 16; ++j) {
    const int i = base + j;
    const float yi = s_y[i] * scale;
    const float yn = s_y[(i + 1) & (T_ - 1)] * scale;
    float out = yi * fminf(2.0f / (1.0f + yi + yn), 1.0f);
    if (i == 0) out = 0.f;
    yo[i] = out;
  }
}

// ---------------------------------------------------------------------------
// Kernel 3: lifted = L2norm(dense @ W_lift + b_lift), dense = [x | sal,tpos,cums]
// WMMA over K=128 (x part) + 3 rank-1 VALU fixups for the topology features.
// ---------------------------------------------------------------------------
__global__ __launch_bounds__(256) void k_lift(
    const float* __restrict__ x, const short* __restrict__ WLp,
    const float* __restrict__ W_lift, const float* __restrict__ b_lift,
    const float* __restrict__ sal, const float* __restrict__ cums,
    float* __restrict__ lifted)
{
  const int wave = threadIdx.x >> 5;
  const int lane = threadIdx.x & 31;
  const int tile = blockIdx.x * 8 + wave;
  const int row0 = tile * 16;
  const int m    = lane & 15;
  const int hi   = lane >> 4;

  v16bf a[4];
  const float* xr = x + (size_t)(row0 + m) * D_;
  __builtin_prefetch(xr + 16 * D_, 0, 0);
#pragma unroll
  for (int ks = 0; ks < 4; ++ks) {
    v16s s;
#pragma unroll
    for (int v = 0; v < 8; ++v) {
      int kb = ks * 32 + ((v & 4) ? 16 : 0) + hi * 8 + (v & 3) * 2;
      s[2 * v]     = f2bf_s(xr[kb]);
      s[2 * v + 1] = f2bf_s(xr[kb + 1]);
    }
    a[ks] = __builtin_bit_cast(v16bf, s);
  }

  float sv[8], cv[8], tv[8];
#pragma unroll
  for (int r = 0; r < 8; ++r) {
    const int g = row0 + r + 8 * hi;
    sv[r] = sal[g];
    cv[r] = cums[g];
    tv[r] = (float)(g & (T_ - 1)) * (1.0f / (float)T_);
  }

  v8f  acc[4];
  float nrm[8] = {0.f, 0.f, 0.f, 0.f, 0.f, 0.f, 0.f, 0.f};

#pragma unroll
  for (int nt = 0; nt < KL_ / 16; ++nt) {      // 4 column tiles of k=64
    const int col = nt * 16 + m;
    v8f c = {};
#pragma unroll
    for (int ks = 0; ks < 4; ++ks) {
      const v16bf bm = load_b_frag(WLp, nt * 4 + ks, lane);
      c = __builtin_amdgcn_wmma_f32_16x16x32_bf16(
          false, a[ks], false, bm, (short)0, c, false, false);
    }
    const float w128 = W_lift[(size_t)128 * KL_ + col];
    const float w129 = W_lift[(size_t)129 * KL_ + col];
    const float w130 = W_lift[(size_t)130 * KL_ + col];
    const float bl   = b_lift[col];
#pragma unroll
    for (int r = 0; r < 8; ++r) {
      const float t = c[r] + sv[r] * w128 + tv[r] * w129 + cv[r] * w130 + bl;
      c[r] = t;
      nrm[r] += t * t;
    }
    acc[nt] = c;
  }

#pragma unroll
  for (int r = 0; r < 8; ++r)
    for (int msk = 8; msk >= 1; msk >>= 1) nrm[r] += __shfl_xor(nrm[r], msk, 32);

#pragma unroll
  for (int r = 0; r < 8; ++r) nrm[r] = 1.0f / fmaxf(sqrtf(nrm[r]), 1e-6f);

#pragma unroll
  for (int nt = 0; nt < 4; ++nt)
#pragma unroll
    for (int r = 0; r < 8; ++r) {
      const int g = row0 + r + 8 * hi;
      lifted[(size_t)g * KL_ + nt * 16 + m] = acc[nt][r] * nrm[r];
    }
}

// ---------------------------------------------------------------------------
// Kernel 4: top-64 indices of y_star per row (descending, stable ties ->
// lowest index first, matching jax.lax.top_k). Iterative argmax in LDS.
// ---------------------------------------------------------------------------
__global__ __launch_bounds__(256) void k_topk(
    const float* __restrict__ ystar, int* __restrict__ topidx)
{
  __shared__ float s_y[T_];
  __shared__ float s_v[256];
  __shared__ int   s_i[256];

  const int b = blockIdx.x, tid = threadIdx.x;
  const float* yr = ystar + (size_t)b * T_;
  for (int i = tid; i < T_; i += 256) s_y[i] = yr[i];
  __syncthreads();

  for (int sel = 0; sel < KEFF; ++sel) {
    float bv = -3.402823466e38f;
    int   bi = 0;
#pragma unroll
    for (int j = 0; j < 16; ++j) {
      const int i = tid * 16 + j;
      const float v = s_y[i];
      if (v > bv) { bv = v; bi = i; }          // ascending scan => lowest-idx tie
    }
    s_v[tid] = bv; s_i[tid] = bi;
    __syncthreads();
    for (int off = 128; off >= 1; off >>= 1) {
      if (tid < off) {
        const float ov = s_v[tid + off];
        const int   oi = s_i[tid + off];
        if (ov > s_v[tid] || (ov == s_v[tid] && oi < s_i[tid])) {
          s_v[tid] = ov; s_i[tid] = oi;
        }
      }
      __syncthreads();
    }
    if (tid == 0) {
      topidx[b * KEFF + sel] = s_i[0];
      s_y[s_i[0]] = -3.402823466e38f;
    }
    __syncthreads();
  }
}

// ---------------------------------------------------------------------------
// Kernel 5: tokens[b,n,:] = lifted[b, idx[b,n], :] @ W_proj + b_proj
// Gather fused into the WMMA A-operand load. K=64 -> 2 bf16 WMMA steps.
// ---------------------------------------------------------------------------
__global__ __launch_bounds__(256) void k_project(
    const float* __restrict__ lifted, const int* __restrict__ topidx,
    const short* __restrict__ WPp, const float* __restrict__ b_proj,
    float* __restrict__ tokens)
{
  const int wave = threadIdx.x >> 5;
  const int lane = threadIdx.x & 31;
  const int tile = blockIdx.x * 8 + wave;      // 8192 = (32*4 Mtiles)*(64 Ntiles)
  const int nt   = tile & 63;
  const int bt   = tile >> 6;
  const int b    = bt >> 2;
  const int mt   = bt & 3;
  const int m    = lane & 15;
  const int hi   = lane >> 4;

  // gather: A row comes from the top-idx anchor row of `lifted`
  const int am = mt * 16 + m;
  const int g  = b * T_ + topidx[b * KEFF + am];
  const float* lr = lifted + (size_t)g * KL_;

  v16bf a[2];
#pragma unroll
  for (int ks = 0; ks < 2; ++ks) {
    v16s s;
#pragma unroll
    for (int v = 0; v < 8; ++v) {
      int kb = ks * 32 + ((v & 4) ? 16 : 0) + hi * 8 + (v & 3) * 2;
      s[2 * v]     = f2bf_s(lr[kb]);
      s[2 * v + 1] = f2bf_s(lr[kb + 1]);
    }
    a[ks] = __builtin_bit_cast(v16bf, s);
  }

  const int col = nt * 16 + m;
  v8f acc = {};
#pragma unroll
  for (int ks = 0; ks < 2; ++ks) {
    const v16bf bm = load_b_frag(WPp, nt * 2 + ks, lane);
    acc = __builtin_amdgcn_wmma_f32_16x16x32_bf16(
        false, a[ks], false, bm, (short)0, acc, false, false);
  }

  const float bp = b_proj[col];
#pragma unroll
  for (int r = 0; r < 8; ++r) {
    const int mr = mt * 16 + r + 8 * hi;
    tokens[((size_t)b * KEFF + mr) * DM_ + col] = acc[r] + bp;
  }
}

// ---------------------------------------------------------------------------
// Host-side launcher
// inputs: 0=x 1=W1 2=b1 3=w_e(unused) 4=w_s 5=W_lift 6=b_lift 7=W_proj
//         8=b_proj 9=log_temperature
// d_out:  tokens [32,64,1024] f32  ||  y_star [32,4096] f32
// ---------------------------------------------------------------------------
extern "C" void kernel_launch(void* const* d_in, const int* in_sizes, int n_in,
                              void* d_out, int out_size, void* d_ws, size_t ws_size,
                              hipStream_t stream) {
  const float* x      = (const float*)d_in[0];
  const float* W1     = (const float*)d_in[1];
  const float* b1     = (const float*)d_in[2];
  const float* w_s    = (const float*)d_in[4];
  const float* W_lift = (const float*)d_in[5];
  const float* b_lift = (const float*)d_in[6];
  const float* W_proj = (const float*)d_in[7];
  const float* b_proj = (const float*)d_in[8];
  const float* lt     = (const float*)d_in[9];

  float* tokens = (float*)d_out;                           // 32*64*1024
  float* ystar  = (float*)d_out + (size_t)B_ * KEFF * DM_; // 32*4096

  // workspace (floats): sal[131072] | cums[131072] | lifted[8388608] |
  //   topidx[2048 int] | W1p[16384] | WLp[4096] | WPp[32768]  (~34.8 MB)
  float* ws     = (float*)d_ws;
  float* sal    = ws;
  float* cums   = ws + 131072;
  float* lifted = ws + 262144;
  int*   topidx = (int*)(ws + 8650752);
  short* W1p    = (short*)(ws + 8652800);   // byte offset 32-aligned
  short* WLp    = (short*)(ws + 8669184);
  short* WPp    = (short*)(ws + 8673280);

  // 1) pack weights into WMMA B-fragment bf16 layout (once per launch)
  k_pack_b<<<(D_ * H_  + 255) / 256, 256, 0, stream>>>(W1,     W1p, D_, H_);
  k_pack_b<<<(D_ * KL_ + 255) / 256, 256, 0, stream>>>(W_lift, WLp, D_, KL_);
  k_pack_b<<<(KL_ * DM_ + 255) / 256, 256, 0, stream>>>(W_proj, WPp, KL_, DM_);

  const int tiles = (B_ * T_) / 16;   // 8192 wave-tiles
  k_saliency<<<tiles / 8, 256, 0, stream>>>(x, W1p, b1, w_s, sal);
  k_selector<<<B_, 256, 0, stream>>>(sal, lt, cums, ystar);
  k_lift    <<<tiles / 8, 256, 0, stream>>>(x, WLp, W_lift, b_lift, sal, cums, lifted);
  k_topk    <<<B_, 256, 0, stream>>>(ystar, topidx);
  k_project <<<tiles / 8, 256, 0, stream>>>(lifted, topidx, WPp, b_proj, tokens);
}
